// Reconstruction_80281528696887
// MI455X (gfx1250) — matrix operand order repaired and checked by reading, not compiled, not run on hardware
//
#include <hip/hip_runtime.h>
#include <hip/hip_bf16.h>

// ---------------- problem constants ----------------
#define AA      512
#define A_USE   500
#define VV      256
#define UU      512
#define DD      192
#define HH      192
#define WW      192

#define M_ROWS  (A_USE * VV)      // 128000 GEMM rows
#define KDIM    UU                // 512
#define NDIM    UU                // 512

typedef __attribute__((ext_vector_type(16))) __bf16 v16bf;
typedef __attribute__((ext_vector_type(8)))  float  v8f;

// ---- f32 <-> bf16 with explicit round-to-nearest-even
__device__ __forceinline__ unsigned short f2bf(float x) {
  unsigned u = __float_as_uint(x);
  unsigned r = (u + 0x7FFFu + ((u >> 16) & 1u)) >> 16;
  return (unsigned short)r;
}
__device__ __forceinline__ float bf2f(unsigned short b) {
  return __uint_as_float(((unsigned)b) << 16);
}

// ---------------------------------------------------------------------------
// Kernel 1: spatial filter kernel  h[j] = (1/U) * sum_k filt[k]*cos(2*pi*k*j/U)
// (ortho fft * filt * ortho ifft == circular convolution with real/even h)
// ---------------------------------------------------------------------------
__global__ void build_h_kernel(const float* __restrict__ filt, float* __restrict__ h) {
  int j = threadIdx.x;               // 0..511
  float s = 0.0f;
  const float c0 = 6.2831853071795864769f / (float)UU;
  for (int k = 0; k < UU; ++k) {
    s += filt[k] * __cosf(c0 * (float)(k * j));
  }
  h[j] = s / (float)UU;
}

// ---------------------------------------------------------------------------
// Kernel 2: circulant matrix Circ[j][u] = h[(u-j) mod U], split into bf16 hi/lo
// ---------------------------------------------------------------------------
__global__ void build_circ_kernel(const float* __restrict__ h,
                                  unsigned short* __restrict__ Chi,
                                  unsigned short* __restrict__ Clo) {
  int j = blockIdx.x;                // K index
  int u = threadIdx.x;               // N index
  float c  = h[(u - j) & (UU - 1)];
  unsigned short hi = f2bf(c);
  unsigned short lo = f2bf(c - bf2f(hi));
  Chi[u * KDIM + j] = hi;
  Clo[u * KDIM + j] = lo;
}

// ---------------------------------------------------------------------------
// Kernel 3: filtering GEMM  Y[M_ROWS,512] = X[M_ROWS,512] * Circ[512,512]
// bf16 WMMA 16x16x32, error-compensated split (hi*hi + hi*lo + lo*hi).
// All 8 waves of a block share tileM -> A panel converted once into LDS.
// Inner loop: 2x2 ds_load_b128 (A) + 2x2 global b128 (B) + 3 WMMA.
// ---------------------------------------------------------------------------
union AFrag { v16bf v; uint4 q[2]; };

__global__ __launch_bounds__(256)
void filter_gemm_kernel(const float* __restrict__ X,
                        const unsigned short* __restrict__ Chi,
                        const unsigned short* __restrict__ Clo,
                        float* __restrict__ Y) {
  __shared__ __align__(16) unsigned short Ahi[16 * KDIM];   // 16 KB
  __shared__ __align__(16) unsigned short Alo[16 * KDIM];   // 16 KB

  const int tid   = threadIdx.x;
  const int tile0 = blockIdx.x * 8;                // 8 tiles per block, same tileM
  const int tileM = tile0 >> 5;                    // / (NDIM/16)
  const size_t rowbase = (size_t)tileM * 16 * KDIM;

  // ---- cooperative f32 -> bf16 hi/lo conversion of the shared A panel
  for (int i = tid; i < 16 * KDIM; i += 256) {
    float xv = X[rowbase + i];                     // 16 consecutive rows, contiguous
    unsigned short hi = f2bf(xv);
    Ahi[i] = hi;
    Alo[i] = f2bf(xv - bf2f(hi));
  }
  __syncthreads();

  const int lane   = tid & 31;
  const int wave   = tid >> 5;
  const int tileN  = (tile0 + wave) & 31;
  const int n0     = tileN * 16;
  const int mlocal = lane & 15;
  const int half   = lane >> 4;

  const unsigned short* __restrict__ arh = &Ahi[mlocal * KDIM + 8 * half];
  const unsigned short* __restrict__ arl = &Alo[mlocal * KDIM + 8 * half];

  v8f acc = {};

  for (int kb = 0; kb < KDIM; kb += 32) {
    // A fragments: dwords 0-3 = K contiguous [kb+8h, kb+8h+7],
    //              dwords 4-7 = K contiguous [kb+16+8h, kb+23+8h]  (ISA table)
    AFrag ahi, alo;
    ahi.q[0] = *(const uint4*)(arh + kb);
    ahi.q[1] = *(const uint4*)(arh + kb + 16);
    alo.q[0] = *(const uint4*)(arl + kb);
    alo.q[1] = *(const uint4*)(arl + kb + 16);
    // B fragments: n = lane%16, same K order as A (Circ stored transposed, n-major)
    AFrag bhi, blo;
    const size_t boff = (size_t)(n0 + mlocal) * KDIM + kb + 8 * half;
    bhi.q[0] = *(const uint4*)(Chi + boff);
    bhi.q[1] = *(const uint4*)(Chi + boff + 16);
    blo.q[0] = *(const uint4*)(Clo + boff);
    blo.q[1] = *(const uint4*)(Clo + boff + 16);

    acc = __builtin_amdgcn_wmma_f32_16x16x32_bf16(false, ahi.v, false, bhi.v,
                                                  (short)0, acc, false, false);
    acc = __builtin_amdgcn_wmma_f32_16x16x32_bf16(false, ahi.v, false, blo.v,
                                                  (short)0, acc, false, false);
    acc = __builtin_amdgcn_wmma_f32_16x16x32_bf16(false, alo.v, false, bhi.v,
                                                  (short)0, acc, false, false);
  }

  // ---- store C/D: VGPR r -> M = r + 8*half, N = lane%16
#pragma unroll
  for (int r = 0; r < 8; ++r) {
    int m = r + 8 * half;
    Y[(size_t)(tileM * 16 + m) * NDIM + n0 + mlocal] = acc[r];
  }
}

// ---------------------------------------------------------------------------
// Kernel 4: voxel-driven FDK backprojection.
// Thread = (ix,iy), block = 16x16 xy tile x 16-deep z chunk.
// p = P*[x,y,z,1] affine in z -> 3 adds per z (VOX = 1).  v_rcp_f32 for 1/w.
// ---------------------------------------------------------------------------
__global__ __launch_bounds__(256)
void backproject_kernel(const float* __restrict__ Y,
                        const float* __restrict__ mats,
                        float* __restrict__ out) {
  const int ix = blockIdx.x * 16 + threadIdx.x;
  const int iy = blockIdx.y * 16 + threadIdx.y;
  const int z0 = blockIdx.z * 16;

  const float x  = (float)ix - (float)(WW - 1) * 0.5f;
  const float y  = (float)iy - (float)(HH - 1) * 0.5f;
  const float zb = (float)z0 - (float)(DD - 1) * 0.5f;

  float acc[16];
#pragma unroll
  for (int i = 0; i < 16; ++i) acc[i] = 0.0f;

  for (int a = 0; a < A_USE; ++a) {
    const float* __restrict__ P = mats + a * 12;   // uniform -> scalar loads
    float p0 = P[0] * x + P[1] * y + P[2]  * zb + P[3];
    float p1 = P[4] * x + P[5] * y + P[6]  * zb + P[7];
    float p2 = P[8] * x + P[9] * y + P[10] * zb + P[11];
    const float dz0 = P[2], dz1 = P[6], dz2 = P[10];
    const float* __restrict__ proj = Y + (size_t)a * (VV * UU);

    if (a + 1 < A_USE)   // lowers to global_prefetch_b8
      __builtin_prefetch(Y + (size_t)(a + 1) * (VV * UU) + (size_t)threadIdx.y * UU, 0, 1);

#pragma unroll 4
    for (int iz = 0; iz < 16; ++iz) {
      const float w = p2;
      const float invw = __builtin_amdgcn_rcpf(w);   // v_rcp_f32
      const float u = p0 * invw;
      const float v = p1 * invw;
      const float u0 = floorf(u);
      const float v0 = floorf(v);
      const float fu = u - u0;
      const float fv = v - v0;
      const int iu = (int)u0;
      const int iv = (int)v0;
      const bool valid = (iu >= 0) & (iu < UU - 1) & (iv >= 0) & (iv < VV - 1) & (w > 0.0f);
      const int iuc = min(max(iu, 0), UU - 2);
      const int ivc = min(max(iv, 0), VV - 2);
      const float* g = proj + (size_t)ivc * UU + iuc;
      const float g00 = g[0];
      const float g01 = g[1];
      const float g10 = g[UU];
      const float g11 = g[UU + 1];
      float val = (1.0f - fv) * ((1.0f - fu) * g00 + fu * g01)
                +         fv  * ((1.0f - fu) * g10 + fu * g11);
      acc[iz] += valid ? (val * invw * invw) : 0.0f;
      p0 += dz0; p1 += dz1; p2 += dz2;
    }
  }

#pragma unroll
  for (int iz = 0; iz < 16; ++iz) {
    out[(size_t)(z0 + iz) * (HH * WW) + (size_t)iy * WW + ix] = acc[iz];
  }
}

// ---------------------------------------------------------------------------
extern "C" void kernel_launch(void* const* d_in, const int* in_sizes, int n_in,
                              void* d_out, int out_size, void* d_ws, size_t ws_size,
                              hipStream_t stream) {
  const float* sino = (const float*)d_in[0];   // [1, 512, 256, 512] f32
  const float* filt = (const float*)d_in[1];   // [512] f32
  const float* mats = (const float*)d_in[2];   // [500, 3, 4] f32
  float* out = (float*)d_out;                  // [1, 192, 192, 192] f32

  // workspace layout
  char* ws = (char*)d_ws;
  float*          h   = (float*)ws;                                  // 2 KB (pad 4 KB)
  unsigned short* Chi = (unsigned short*)(ws + 4096);                // 512 KB
  unsigned short* Clo = Chi + (size_t)KDIM * NDIM;                   // 512 KB
  float*          Yf  = (float*)(ws + 4096 + 2u * KDIM * NDIM * 2u); // 250 MB

  build_h_kernel<<<1, UU, 0, stream>>>(filt, h);
  build_circ_kernel<<<KDIM, NDIM, 0, stream>>>(h, Chi, Clo);

  const int tiles = (M_ROWS / 16) * (NDIM / 16);  // 8000 * 32 = 256000 waves
  filter_gemm_kernel<<<tiles / 8, 256, 0, stream>>>(sino, Chi, Clo, Yf);

  dim3 bgrid(WW / 16, HH / 16, DD / 16);
  dim3 bblk(16, 16, 1);
  backproject_kernel<<<bgrid, bblk, 0, stream>>>(Yf, mats, out);
}
